// SharedBrain_927712936574
// MI455X (gfx1250) — matrix operand order repaired and checked by reading, not compiled, not run on hardware
//
#include <hip/hip_runtime.h>

// ===========================================================================
// SharedBrain fused forward for gfx1250 (MI455X).
//  - All GEMMs via v_wmma_f32_16x16x32_bf16 (f32 accumulate).
//  - One wave32 handles a 16-row batch tile; activations staged in LDS bf16.
//  - Weights pre-packed once into d_ws as bf16 [Npad][Kpad] (B-fragment = one
//    contiguous 32B load per lane).  fm MLP is dead code; ctx == 1/16 exactly.
//  - Biases preloaded per-GEMM before the WMMA loop (overlap loads w/ matrix
//    ops); h_next f32 buffer aliased onto the f32 scratch to cut LDS ~17%.
// Input flattening assumed (setup_inputs dict order, tuples as (w,b)):
//  0 z  1 task_ids  2 task_emb  3 r1w 4 r1b 5 r2w 6 r2b  7 expW1 8 expb1
//  9 expW2 10 expb2  11-16 fm(unused)  17-22 nov  23-28 why  29-34 iff
//  35-40 think  41-46 outn
// ===========================================================================

typedef __bf16 bf16_t;
typedef __bf16 v16bf __attribute__((ext_vector_type(16)));
typedef __bf16 v8bf  __attribute__((ext_vector_type(8)));
typedef float  v8f   __attribute__((ext_vector_type(8)));

#define KPAD 160        // A-staging pitch (max K = 2*D = 160)
#define NW   2          // waves per workgroup
#define BATCH 65536
#define DM   80

#define WB() __builtin_amdgcn_wave_barrier()

__device__ __forceinline__ v8f wmma_bf16(v16bf a, v16bf b, v8f c) {
  return __builtin_amdgcn_wmma_f32_16x16x32_bf16(false, a, false, b,
                                                 (short)0, c, false, false);
}

// A fragment (16x32 bf16) from LDS staging buffer, pitch KPAD.
// lanes 0-15: row m=lane, elems: K = kb*32 + 0..15
// lanes 16-31: row m=lane-16, elems: K = kb*32 + 16..31
__device__ __forceinline__ v16bf load_A_frag(const bf16_t* Ab, int kb, int lane) {
  int m = lane & 15;
  int kbase = kb * 32 + ((lane & 16) ? 16 : 0);
  const bf16_t* p = Ab + m * KPAD + kbase;
  v8bf lo = *(const v8bf*)(p);
  v8bf hi = *(const v8bf*)(p + 8);
  return __builtin_shufflevector(lo, hi, 0,1,2,3,4,5,6,7,8,9,10,11,12,13,14,15);
}

// B fragment (32x16 bf16) from global pre-packed weights, layout [Npad][Kw].
// lanes 0-15: col n, K = kb*32 + 0..15 ; lanes 16-31: K = kb*32 + 16..31
__device__ __forceinline__ v16bf load_B_frag(const bf16_t* W, int Kw,
                                             int kb, int nb, int lane) {
  int n = nb * 16 + (lane & 15);
  int kbase = kb * 32 + ((lane & 16) ? 16 : 0);
  return *(const v16bf*)(W + n * Kw + kbase);
}

// OUT[16][NB*16] = act(A[16][KB*32] @ W + bias)
// Optional f32 store to out32 (pitch pitchO, cols < Nvalid) and/or bf16
// re-staging to dstA at column dstcol (call sites guarantee NB*16 == N).
template<int KB, int NB>
__device__ __forceinline__ void gemm_tile(const bf16_t* A, const bf16_t* W, int Kw,
                                          const float* bias, int lane, int act,
                                          float* out32, int pitchO,
                                          bf16_t* dstA, int dstcol, int Nvalid) {
  const int nl = lane & 15;
  const int mb = (lane & 16) ? 8 : 0;
  // Preload biases first so the scalar loads overlap the WMMA stream.
  float bv[NB];
#pragma unroll
  for (int nb = 0; nb < NB; ++nb) {
    int n = nb * 16 + nl;
    bv[nb] = (n < Nvalid) ? bias[n] : 0.0f;
  }
  v16bf a[KB];
#pragma unroll
  for (int kb = 0; kb < KB; ++kb) a[kb] = load_A_frag(A, kb, lane);
#pragma unroll
  for (int nb = 0; nb < NB; ++nb) {
    v8f acc = {0.f,0.f,0.f,0.f,0.f,0.f,0.f,0.f};
#pragma unroll
    for (int kb = 0; kb < KB; ++kb)
      acc = wmma_bf16(a[kb], load_B_frag(W, Kw, kb, nb, lane), acc);
    int n = nb * 16 + nl;
#pragma unroll
    for (int v = 0; v < 8; ++v) {
      float x = acc[v] + bv[nb];
      if (act == 1) x = fmaxf(x, 0.0f);
      int m = mb + v;
      if (out32 && n < Nvalid) out32[m * pitchO + n] = x;
      if (dstA) dstA[m * KPAD + dstcol + n] = (bf16_t)x;
    }
  }
}

struct KParams {
  const float* z; const int* tid; const float* temb;
  const float *r1b, *r2b, *eb1, *eb2;
  const float *nov0b, *nov1b, *nov2w, *nov2b;
  const float *why0b, *why1b, *why2b;
  const float *iff0b, *iff1b, *iff2w, *iff2b;
  const float *think0b, *think1b, *think2b;
  const float *outn0b, *outn1b, *outn2b;
  const bf16_t *r1w, *r2w, *eW1, *eW2, *nov0w, *nov1w;
  const bf16_t *why0w, *why1w, *why2w, *iff0w, *iff1w;
  const bf16_t *think0w, *think1w, *think2w, *outn0w, *outn1w, *outn2w;
  float* out;
};

__global__ __launch_bounds__(NW * 32)
void brain_kernel(KParams P) {
  __shared__ bf16_t Ssh[NW][16 * KPAD];
  __shared__ bf16_t Tsh[NW][16 * KPAD];
  __shared__ float  Xsh[NW][16 * 128];   // f32 scratch; also aliases h_next
  __shared__ float  Hsh[NW][16 * DM];
  __shared__ float  MAsh[NW][256];
  __shared__ float  MBsh[NW][256];
  __shared__ float  Rnov[NW][16], Rifq[NW][16], Rcur[NW][16];

  const int lane = threadIdx.x & 31;
  const int wv   = threadIdx.x >> 5;
  const int row0 = (blockIdx.x * NW + wv) * 16;

  bf16_t* S = Ssh[wv]; bf16_t* T = Tsh[wv];
  float* X = Xsh[wv];  float* H = Hsh[wv];
  float* HN = Xsh[wv];                  // h_next aliases X (never live together)
  float* MA = MAsh[wv]; float* MB = MBsh[wv];

  // ---- build h0 = [z, task_emb[task]] ; stage bf16 (pad K to 96 with zeros)
  for (int idx = lane; idx < 16 * DM; idx += 32) {
    int r = idx / DM, c = idx % DM, row = row0 + r;
    float v = (c < 64) ? P.z[row * 64 + c] : P.temb[P.tid[row] * 16 + (c - 64)];
    H[r * DM + c] = v;
    S[r * KPAD + c] = (bf16_t)v;
  }
  for (int idx = lane; idx < 16 * 16; idx += 32) {
    int r = idx / 16, c = 80 + (idx & 15);
    S[r * KPAD + c] = (bf16_t)0.f;
  }
  if (lane < 16) Rcur[wv][lane] = 0.f;
  WB();

  const int nl = lane & 15;
  const int mb = (lane & 16) ? 8 : 0;

#pragma unroll 1
  for (int st = 0; st < 2; ++st) {
    // ---- routers: scores1 | scores2 into X (pitch 128)
    gemm_tile<3,1>(S, P.r1w, 96, P.r1b, lane, 0, X,      128, nullptr, 0, 16);
    gemm_tile<3,1>(S, P.r2w, 96, P.r2b, lane, 0, X + 16, 128, nullptr, 0, 16);
    WB();
    {  // lane l<16: router1 row l ; lane>=16: router2 row l-16
      int m = lane & 15;
      const float* sc = X + m * 128 + ((lane & 16) ? 16 : 0);
      float v0 = -3.4e38f, v1 = -3.4e38f; int i0 = 0, i1 = 0;
      for (int e = 0; e < 16; ++e) {
        float s = sc[e];
        if (s > v0)      { v1 = v0; i1 = i0; v0 = s; i0 = e; }
        else if (s > v1) { v1 = s;  i1 = e; }
      }
      float p0 = 1.f / (1.f + __expf(v1 - v0));
      float p1 = 1.f - p0;
      float* MH = (lane & 16) ? MB : MA;
      for (int e = 0; e < 16; ++e)
        MH[m * 16 + e] = (e == i0) ? p0 : ((e == i1) ? p1 : 0.f);
    }
    WB();
    // ---- novelty MLP on h : 80->80 relu -> 80->40 relu -> 40->1 sigmoid
    gemm_tile<3,5>(S, P.nov0w, 96, P.nov0b, lane, 1, nullptr, 0, T, 0, 80);
    for (int idx = lane; idx < 16 * 16; idx += 32) {
      int r = idx / 16, c = 80 + (idx & 15);
      T[r * KPAD + c] = (bf16_t)0.f;
    }
    WB();
    gemm_tile<3,3>(T, P.nov1w, 96, P.nov1b, lane, 1, X, 48, nullptr, 0, 40);
    WB();
    if (lane < 16) {
      float s = P.nov2b[0];
      for (int k = 0; k < 40; ++k) s += X[lane * 48 + k] * P.nov2w[k];
      Rnov[wv][lane] = 1.f / (1.f + __expf(-s));
    }
    WB();
    // ---- dense expert sweep, mix-weighted accumulation in registers
    float hnf[5][8];
#pragma unroll
    for (int nb = 0; nb < 5; ++nb)
#pragma unroll
      for (int v = 0; v < 8; ++v) hnf[nb][v] = 0.f;
#pragma unroll 1
    for (int e = 0; e < 16; ++e) {
      gemm_tile<3,4>(S, P.eW1 + e * (64 * 96), 96, P.eb1 + e * 64,
                     lane, 1, nullptr, 0, T, 0, 64);
      WB();
      float mc[8];
#pragma unroll
      for (int v = 0; v < 8; ++v)
        mc[v] = 0.5f * (MA[(mb + v) * 16 + e] + MB[(mb + v) * 16 + e]);
      // Preload layer-2 biases so they overlap the WMMA stream.
      float bv2[5];
#pragma unroll
      for (int nb = 0; nb < 5; ++nb) bv2[nb] = P.eb2[e * 80 + nb * 16 + nl];
      v16bf a0 = load_A_frag(T, 0, lane);
      v16bf a1 = load_A_frag(T, 1, lane);
      const bf16_t* W2 = P.eW2 + e * (80 * 64);
#pragma unroll
      for (int nb = 0; nb < 5; ++nb) {
        v8f acc = {0.f,0.f,0.f,0.f,0.f,0.f,0.f,0.f};
        acc = wmma_bf16(a0, load_B_frag(W2, 64, 0, nb, lane), acc);
        acc = wmma_bf16(a1, load_B_frag(W2, 64, 1, nb, lane), acc);
#pragma unroll
        for (int v = 0; v < 8; ++v)
          hnf[nb][v] += mc[v] * fmaxf(acc[v] + bv2[nb], 0.f);
      }
      WB();
    }
    // ---- commit h_next (f32 into HN(=X) + bf16 restage of S cols 0..79)
#pragma unroll
    for (int nb = 0; nb < 5; ++nb)
#pragma unroll
      for (int v = 0; v < 8; ++v) {
        int n = nb * 16 + nl, m = mb + v;
        HN[m * DM + n] = hnf[nb][v];
        S[m * KPAD + n] = (bf16_t)hnf[nb][v];
      }
    WB();
    // ---- curiosity
    if (lane < 16) {
      float s = 0.f;
      for (int c = 0; c < DM; ++c) {
        float d = HN[lane * DM + c] - H[lane * DM + c];
        s += d * d;
      }
      float err  = sqrtf(s);
      float errn = 1.f / (1.f + __expf(0.5f - err));
      float cur  = fminf(fmaxf(0.4f * errn + 0.6f * Rnov[wv][lane], 0.1f), 0.9f);
      Rcur[wv][lane] += cur;
    }
    WB();
    for (int idx = lane; idx < 16 * DM; idx += 32) H[idx] = HN[idx];
    WB();
  }

  // ===================== self-awareness (ctx == 1/16) ======================
  // hwc = [h, h/16] -> S cols 0..159
  for (int idx = lane; idx < 16 * DM; idx += 32) {
    int r = idx / DM, c = idx % DM;
    S[r * KPAD + 80 + c] = (bf16_t)(H[r * DM + c] * 0.0625f);
  }
  WB();
  gemm_tile<5,8>(S, P.why0w, 160, P.why0b, lane, 1, nullptr, 0, T, 0, 128);
  WB();
  gemm_tile<4,8>(T, P.why1w, 128, P.why1b, lane, 1, nullptr, 0, T, 0, 128);
  WB();
  // why output -> S cols 80..143 (h stays in cols 0..79) ; zero 144..159
  gemm_tile<4,4>(T, P.why2w, 128, P.why2b, lane, 0, nullptr, 0, S, 80, 64);
  for (int idx = lane; idx < 16 * 16; idx += 32) {
    int r = idx / 16, c = 144 + (idx & 15);
    S[r * KPAD + c] = (bf16_t)0.f;
  }
  WB();
  // iff: 144->128 relu -> 128->64 relu -> 64->1 sigmoid
  gemm_tile<5,8>(S, P.iff0w, 160, P.iff0b, lane, 1, nullptr, 0, T, 0, 128);
  WB();
  gemm_tile<4,4>(T, P.iff1w, 128, P.iff1b, lane, 1, X, 64, nullptr, 0, 64);
  WB();
  if (lane < 16) {
    float s = P.iff2b[0];
    for (int k = 0; k < 64; ++k) s += X[lane * 64 + k] * P.iff2w[k];
    Rifq[wv][lane] = 1.f / (1.f + __expf(-s));
  }
  WB();
  // think: 144->128 relu -> 128->128 relu -> 128->80
  gemm_tile<5,8>(S, P.think0w, 160, P.think0b, lane, 1, nullptr, 0, T, 0, 128);
  WB();
  gemm_tile<4,8>(T, P.think1w, 128, P.think1b, lane, 1, nullptr, 0, T, 0, 128);
  WB();
  gemm_tile<4,5>(T, P.think2w, 128, P.think2b, lane, 0, X, 80, nullptr, 0, 80);
  WB();
  // blended = h + ifq*think -> S cols 80..159 (outn input = [h, blended])
  for (int idx = lane; idx < 16 * DM; idx += 32) {
    int r = idx / DM, c = idx % DM;
    S[r * KPAD + 80 + c] = (bf16_t)(H[r * DM + c] + Rifq[wv][r] * X[r * DM + c]);
  }
  WB();
  gemm_tile<5,8>(S, P.outn0w, 160, P.outn0b, lane, 1, nullptr, 0, T, 0, 128);
  WB();
  gemm_tile<4,8>(T, P.outn1w, 128, P.outn1b, lane, 1, nullptr, 0, T, 0, 128);
  WB();
  gemm_tile<4,5>(T, P.outn2w, 128, P.outn2b, lane, 0, X, 80, nullptr, 0, 80);
  WB();
  // ---- h_final = 0.7h + 0.3 h_ref ; avg_cur
  for (int idx = lane; idx < 16 * DM; idx += 32) {
    int r = idx / DM, c = idx % DM;
    P.out[(row0 + r) * DM + c] = 0.7f * H[r * DM + c] + 0.3f * X[r * DM + c];
  }
  if (lane < 16) P.out[BATCH * DM + row0 + lane] = 0.5f * Rcur[wv][lane];
}

// ---------------------------------------------------------------------------
// Weight pre-pack: f32 [nmat][K][N] row-major -> bf16 [nmat][Npad][Kpad]
// (n-major so each B-fragment lane load is one contiguous 32B read), zero-pad.
// ---------------------------------------------------------------------------
__global__ void convert_wt(const float* __restrict__ src, bf16_t* __restrict__ dst,
                           int K, int N, int Kp, int Np, int nmat) {
  long long per = (long long)Np * Kp;
  long long total = per * nmat;
  for (long long idx = (long long)blockIdx.x * blockDim.x + threadIdx.x;
       idx < total; idx += (long long)gridDim.x * blockDim.x) {
    int e = (int)(idx / per);
    int r = (int)(idx % per);
    int n = r / Kp, k = r % Kp;
    float v = (n < N && k < K) ? src[(long long)e * K * N + (long long)k * N + n] : 0.f;
    dst[idx] = (bf16_t)v;
  }
}

extern "C" void kernel_launch(void* const* d_in, const int* in_sizes, int n_in,
                              void* d_out, int out_size, void* d_ws, size_t ws_size,
                              hipStream_t stream) {
  bf16_t* ws = (bf16_t*)d_ws;
  // bf16-element offsets into d_ws (total 363520 elem = 727,040 bytes)
  const size_t off_r1 = 0, off_r2 = 1536, off_eW1 = 3072, off_eW2 = 101376,
               off_nov0 = 183296, off_nov1 = 190976, off_why0 = 195584,
               off_why1 = 216064, off_why2 = 232448, off_iff0 = 240640,
               off_iff1 = 261120, off_think0 = 269312, off_think1 = 289792,
               off_think2 = 306176, off_outn0 = 316416, off_outn1 = 336896,
               off_outn2 = 353280;

  auto F = [&](int i) { return (const float*)d_in[i]; };
  auto conv = [&](int i, size_t off, int K, int N, int Kp, int Np, int nm) {
    long long total = (long long)nm * Np * Kp;
    int grid = (int)((total + 255) / 256);
    if (grid > 2048) grid = 2048;
    convert_wt<<<grid, 256, 0, stream>>>(F(i), ws + off, K, N, Kp, Np, nm);
  };
  conv(3,  off_r1,    80, 16,  96, 16,  1);
  conv(5,  off_r2,    80, 16,  96, 16,  1);
  conv(7,  off_eW1,   80, 64,  96, 64,  16);
  conv(9,  off_eW2,   64, 80,  64, 80,  16);
  conv(17, off_nov0,  80, 80,  96, 80,  1);
  conv(19, off_nov1,  80, 40,  96, 48,  1);
  conv(23, off_why0, 160, 128, 160, 128, 1);
  conv(25, off_why1, 128, 128, 128, 128, 1);
  conv(27, off_why2, 128, 64,  128, 64,  1);
  conv(29, off_iff0, 144, 128, 160, 128, 1);
  conv(31, off_iff1, 128, 64,  128, 64,  1);
  conv(35, off_think0, 144, 128, 160, 128, 1);
  conv(37, off_think1, 128, 128, 128, 128, 1);
  conv(39, off_think2, 128, 80,  128, 80,  1);
  conv(41, off_outn0, 160, 128, 160, 128, 1);
  conv(43, off_outn1, 128, 128, 128, 128, 1);
  conv(45, off_outn2, 128, 80,  128, 80,  1);

  KParams P;
  P.z = F(0); P.tid = (const int*)d_in[1]; P.temb = F(2);
  P.r1b = F(4);  P.r2b = F(6);  P.eb1 = F(8);  P.eb2 = F(10);
  P.nov0b = F(18); P.nov1b = F(20); P.nov2w = F(21); P.nov2b = F(22);
  P.why0b = F(24); P.why1b = F(26); P.why2b = F(28);
  P.iff0b = F(30); P.iff1b = F(32); P.iff2w = F(33); P.iff2b = F(34);
  P.think0b = F(36); P.think1b = F(38); P.think2b = F(40);
  P.outn0b = F(42); P.outn1b = F(44); P.outn2b = F(46);
  P.r1w = ws + off_r1; P.r2w = ws + off_r2;
  P.eW1 = ws + off_eW1; P.eW2 = ws + off_eW2;
  P.nov0w = ws + off_nov0; P.nov1w = ws + off_nov1;
  P.why0w = ws + off_why0; P.why1w = ws + off_why1; P.why2w = ws + off_why2;
  P.iff0w = ws + off_iff0; P.iff1w = ws + off_iff1;
  P.think0w = ws + off_think0; P.think1w = ws + off_think1; P.think2w = ws + off_think2;
  P.outn0w = ws + off_outn0; P.outn1w = ws + off_outn1; P.outn2w = ws + off_outn2;
  P.out = (float*)d_out;

  const int tiles = BATCH / 16;       // 4096 16-row tiles
  const int blocks = tiles / NW;      // 2048 workgroups of NW waves
  brain_kernel<<<blocks, NW * 32, 0, stream>>>(P);
}
